// lstm_26388279067033
// MI455X (gfx1250) — compile-verified, matrix operand order checked by reading, not run on hardware
//
#include <hip/hip_runtime.h>

// ---------------------------------------------------------------------------
// MI455X (gfx1250) bidirectional-LSTM + MLP, bf16 WMMA pipeline.
// wave32; v_wmma_f32_16x16x32_bf16 for all matmuls; f32 cell state.
// LSTM: U cannot be register-resident (128 VGPRs/wave needed vs ~128 cap at
// 8 waves/SIMD), so B frags intentionally STREAM from the L2-pinned 1MB
// weight table every step (amortized over 2 m-tiles). An asm pointer-launder
// inside the step loop defeats LICM so the compiler cannot hoist the 16 B
// fragments (128 VGPRs) out of the loop and spill them to scratch.
// ---------------------------------------------------------------------------

typedef __bf16 bf16;
typedef __attribute__((ext_vector_type(16))) __bf16 v16bf;
typedef __attribute__((ext_vector_type(8)))  float  v8f;
typedef __attribute__((ext_vector_type(4)))  float  f32x4;

#define B_   4096
#define T_   22
#define V_   5
#define E_   128
#define U_   256
#define G4_  1024      // 4*U
#define F_   128
#define H_   512
#define ZW_  11392     // T_*2*U_ + F_
#define LROWS 32       // batch rows per LSTM workgroup (2 m-tiles)
#define HB_STRIDE 264  // padded bf16 row stride for LDS h buffer (bank-conflict free)

__device__ __forceinline__ float sigm_f(float x) {
    return 1.0f / (1.0f + __expf(-x));
}
__device__ __forceinline__ float tanh_f(float x) {
    x = fminf(fmaxf(x, -15.0f), 15.0f);
    float e = __expf(2.0f * x);
    return (e - 1.0f) / (e + 1.0f);
}

// Load one 16-element bf16 WMMA fragment row-slice: 8 bf16 at p, 8 bf16 at p+16
// (matches the 16-bit A/B VGPR layout: K pairs 0..7 then 16..23 per lane half).
__device__ __forceinline__ v16bf ldfrag(const bf16* p) {
    v16bf r;
    ((f32x4*)&r)[0] = ((const f32x4*)p)[0];
    ((f32x4*)&r)[1] = ((const f32x4*)(p + 16))[0];
    return r;
}

// ---------------------------------------------------------------------------
// Prep: embW[dir][v][g] = emb[v,:] @ W[:,g] + b[g]   (input projection -> LUT)
// ---------------------------------------------------------------------------
__global__ void embw_kernel(const float* __restrict__ emb,
                            const float* __restrict__ Wf, const float* __restrict__ bfv,
                            const float* __restrict__ Wb, const float* __restrict__ bbv,
                            float* __restrict__ embW) {
    int idx = blockIdx.x * 256 + threadIdx.x;
    if (idx >= 2 * V_ * G4_) return;
    int dir = idx / (V_ * G4_);
    int rem = idx % (V_ * G4_);
    int v = rem / G4_, g = rem % G4_;
    const float* W  = dir ? Wb  : Wf;
    const float* bv = dir ? bbv : bfv;
    float s = bv[g];
    #pragma unroll 8
    for (int e = 0; e < E_; ++e) s += emb[v * E_ + e] * W[e * G4_ + g];
    embW[idx] = s;
}

// dst[n*rows + k] = (bf16) src[k*cols + n]   (transpose + f32->bf16)
__global__ void transcvt_kernel(const float* __restrict__ src, bf16* __restrict__ dst,
                                int rows, int cols) {
    int idx = blockIdx.x * 256 + threadIdx.x;
    if (idx >= rows * cols) return;
    int n = idx / rows, k = idx % rows;
    dst[idx] = (bf16)src[(size_t)k * cols + n];
}

__global__ void bio_kernel(const float* __restrict__ bio, bf16* __restrict__ zbuf) {
    int idx = blockIdx.x * 256 + threadIdx.x;
    if (idx >= B_ * F_) return;
    int b = idx / F_, f = idx % F_;
    zbuf[(size_t)b * ZW_ + (T_ * 2 * U_) + f] = (bf16)bio[idx];
}

// ---------------------------------------------------------------------------
// LSTM: one WG = 32 batch rows x one direction, 32 waves (1024 threads).
// Wave w owns z columns [32w, 32w+32) (2 n-tiles) for both m-tiles: per
// k-step it loads 2 B frags (global/L2) + 2 A frags (LDS) and issues 4 WMMAs.
// Gate fusion (incl. the embW LUT add) happens in a per-thread epilogue via a
// 128KB LDS z-stage; f32 cell state lives in the epilogue threads' registers.
// ---------------------------------------------------------------------------
__global__ __launch_bounds__(1024) void lstm_kernel(const int*   __restrict__ seq,
                                                    const float* __restrict__ embW,
                                                    const bf16*  __restrict__ UT,
                                                    bf16*        __restrict__ zout) {
    __shared__ float zs[LROWS * 1024];            // gate pre-activations (f32)
    __shared__ bf16  hbuf[LROWS * HB_STRIDE];     // h (bf16) -> next step A frags

    const int tid  = threadIdx.x;
    const int wave = tid >> 5, lane = tid & 31;
    const int lo   = lane & 15, hi = lane >> 4;
    const int dir  = blockIdx.y;
    const int base = blockIdx.x * LROWS;

    for (int i = tid; i < LROWS * HB_STRIDE; i += 1024) hbuf[i] = (bf16)0.0f;

    // epilogue mapping: thread -> (row em, 8 hidden units at ej)
    const int em = tid >> 5;          // 32 rows (== wave id)
    const int ej = (tid & 31) * 8;    // 8 hidden units
    float c8[8];
    #pragma unroll
    for (int q = 0; q < 8; ++q) c8[q] = 0.0f;

    const bf16*  Ud     = UT + (size_t)dir * (G4_ * U_);
    const float* eW     = embW + dir * (V_ * G4_);
    const int*   seqrow = seq + (base + em) * T_;

    // wave's two B-fragment row bases (columns owned by this wave)
    const bf16* Bn0 = Ud + (size_t)(wave * 32 + lo) * U_;
    const bf16* Bn1 = Ud + (size_t)(wave * 32 + 16 + lo) * U_;

    __syncthreads();

    for (int s = 0; s < T_; ++s) {
        const int t = dir ? (T_ - 1 - s) : s;

        // Defeat LICM: make the B-fragment addresses opaque per iteration so
        // the 16 B frags (128 VGPRs) are never hoisted out and spilled.
        int bko = hi * 8;
        asm volatile("" : "+v"(bko));

        // z = h @ U : K=256 -> 8 k-steps x (2 B frags, 2 A frags, 4 WMMAs).
        v8f acc[2][2];   // [m-tile][n-tile]
        #pragma unroll
        for (int mi = 0; mi < 2; ++mi)
            #pragma unroll
            for (int nl = 0; nl < 2; ++nl)
                #pragma unroll
                for (int r = 0; r < 8; ++r) acc[mi][nl][r] = 0.0f;

        #pragma unroll
        for (int k = 0; k < 8; ++k) {
            const int bkoff = k * 32 + bko;       // B addresses (opaque base)
            const int koff  = k * 32 + hi * 8;    // A addresses (LDS)
            v16bf b0 = ldfrag(Bn0 + bkoff);
            v16bf b1 = ldfrag(Bn1 + bkoff);
            v16bf a0 = ldfrag(&hbuf[lo * HB_STRIDE + koff]);
            v16bf a1 = ldfrag(&hbuf[(16 + lo) * HB_STRIDE + koff]);
            acc[0][0] = __builtin_amdgcn_wmma_f32_16x16x32_bf16(
                false, a0, false, b0, (short)0, acc[0][0], false, false);
            acc[0][1] = __builtin_amdgcn_wmma_f32_16x16x32_bf16(
                false, a0, false, b1, (short)0, acc[0][1], false, false);
            acc[1][0] = __builtin_amdgcn_wmma_f32_16x16x32_bf16(
                false, a1, false, b0, (short)0, acc[1][0], false, false);
            acc[1][1] = __builtin_amdgcn_wmma_f32_16x16x32_bf16(
                false, a1, false, b1, (short)0, acc[1][1], false, false);
        }
        __syncthreads();   // all waves done reading hbuf

        // stage z in LDS
        #pragma unroll
        for (int mi = 0; mi < 2; ++mi)
            #pragma unroll
            for (int nl = 0; nl < 2; ++nl) {
                const int col = wave * 32 + nl * 16 + lo;
                #pragma unroll
                for (int r = 0; r < 8; ++r)
                    zs[(mi * 16 + r + 8 * hi) * 1024 + col] = acc[mi][nl][r];
            }
        __syncthreads();   // zs ready

        // gate fusion: add LUT (xW+b), 8 hidden units/thread, f32 cell state
        const int vtok = seqrow[t];
        const float* ev = eW + vtok * G4_ + ej;
        const float* zr = zs + em * 1024 + ej;
        bf16 h8[8] __attribute__((aligned(16)));
        #pragma unroll
        for (int half = 0; half < 2; ++half) {
            f32x4 zi = *(const f32x4*)&zr[  0 + 4 * half] + *(const f32x4*)&ev[  0 + 4 * half];
            f32x4 zf = *(const f32x4*)&zr[256 + 4 * half] + *(const f32x4*)&ev[256 + 4 * half];
            f32x4 zg = *(const f32x4*)&zr[512 + 4 * half] + *(const f32x4*)&ev[512 + 4 * half];
            f32x4 zo = *(const f32x4*)&zr[768 + 4 * half] + *(const f32x4*)&ev[768 + 4 * half];
            #pragma unroll
            for (int q = 0; q < 4; ++q) {
                float iv = sigm_f(zi[q]);
                float fv = sigm_f(zf[q]);
                float gv = tanh_f(zg[q]);
                float ov = sigm_f(zo[q]);
                float c  = fv * c8[4 * half + q] + iv * gv;
                c8[4 * half + q] = c;
                h8[4 * half + q] = (bf16)(ov * tanh_f(c));
            }
        }
        *(f32x4*)&hbuf[em * HB_STRIDE + ej] = *(const f32x4*)h8;
        *(f32x4*)&zout[(size_t)(base + em) * ZW_ + t * 512 + dir * 256 + ej] =
            *(const f32x4*)h8;
        __syncthreads();   // new h visible before next step's reads
    }
}

// ---------------------------------------------------------------------------
// GEMM: C[M,N] = relu(A[M,K] @ B[K,N] + bias). A bf16 row-major, BT = B
// transposed (n-major) bf16, C bf16. 128x64 tile per WG, 8 waves, 2x2
// C-tiles per wave (4 WMMAs per 4 fragment loads).
// ---------------------------------------------------------------------------
__global__ __launch_bounds__(256) void gemm_kernel(const bf16*  __restrict__ A,
                                                   const bf16*  __restrict__ BT,
                                                   const float* __restrict__ bias,
                                                   bf16*        __restrict__ C,
                                                   int K, int N, int do_relu) {
    const int tid  = threadIdx.x;
    const int wave = tid >> 5, lane = tid & 31;
    const int lo   = lane & 15, hi = lane >> 4;
    const int mw = wave & 3;          // 4 m positions x 2 m-tiles = 128 rows
    const int nw = wave >> 2;         // 2 n positions x 2 n-tiles = 64 cols
    const int mbase = blockIdx.x * 128 + mw * 32;
    const int nbase = blockIdx.y * 64 + nw * 32;

    v8f acc[2][2];
    #pragma unroll
    for (int mi = 0; mi < 2; ++mi)
        #pragma unroll
        for (int ni = 0; ni < 2; ++ni)
            #pragma unroll
            for (int r = 0; r < 8; ++r) acc[mi][ni][r] = 0.0f;

    const bf16* A0 = A  + (size_t)(mbase + lo) * K;
    const bf16* A1 = A  + (size_t)(mbase + 16 + lo) * K;
    const bf16* B0 = BT + (size_t)(nbase + lo) * K;
    const bf16* B1 = BT + (size_t)(nbase + 16 + lo) * K;

    const int kiters = K >> 5;
    for (int k = 0; k < kiters; ++k) {
        const int koff = k * 32 + hi * 8;
        v16bf a0 = ldfrag(A0 + koff);
        v16bf a1 = ldfrag(A1 + koff);
        v16bf b0 = ldfrag(B0 + koff);
        v16bf b1 = ldfrag(B1 + koff);
        acc[0][0] = __builtin_amdgcn_wmma_f32_16x16x32_bf16(
            false, a0, false, b0, (short)0, acc[0][0], false, false);
        acc[0][1] = __builtin_amdgcn_wmma_f32_16x16x32_bf16(
            false, a0, false, b1, (short)0, acc[0][1], false, false);
        acc[1][0] = __builtin_amdgcn_wmma_f32_16x16x32_bf16(
            false, a1, false, b0, (short)0, acc[1][0], false, false);
        acc[1][1] = __builtin_amdgcn_wmma_f32_16x16x32_bf16(
            false, a1, false, b1, (short)0, acc[1][1], false, false);
    }

    #pragma unroll
    for (int mi = 0; mi < 2; ++mi) {
        #pragma unroll
        for (int ni = 0; ni < 2; ++ni) {
            const int n  = nbase + ni * 16 + lo;
            const float bv = bias[n];
            #pragma unroll
            for (int r = 0; r < 8; ++r) {
                const int row = mbase + mi * 16 + r + 8 * hi;
                float v = acc[mi][ni][r] + bv;
                if (do_relu) v = fmaxf(v, 0.0f);
                C[(size_t)row * N + n] = (bf16)v;
            }
        }
    }
}

// ---------------------------------------------------------------------------
// Head: out[b] = C2[b,:] @ Wo + bo   (one wave per row, shuffle reduce)
// ---------------------------------------------------------------------------
__global__ __launch_bounds__(256) void head_kernel(const bf16*  __restrict__ C2,
                                                   const float* __restrict__ Wo,
                                                   const float* __restrict__ bo,
                                                   float* __restrict__ out) {
    const int wave = threadIdx.x >> 5, lane = threadIdx.x & 31;
    const int row  = blockIdx.x * 8 + wave;
    float s = 0.0f;
    #pragma unroll
    for (int q = 0; q < 16; ++q) {
        const int col = lane + 32 * q;
        s += (float)C2[(size_t)row * H_ + col] * Wo[col];
    }
    #pragma unroll
    for (int off = 16; off > 0; off >>= 1) s += __shfl_xor(s, off, 32);
    if (lane == 0) out[row] = s + bo[0];
}

// ---------------------------------------------------------------------------
extern "C" void kernel_launch(void* const* d_in, const int* in_sizes, int n_in,
                              void* d_out, int out_size, void* d_ws, size_t ws_size,
                              hipStream_t stream) {
    const int*   seq = (const int*)  d_in[0];
    const float* bio = (const float*)d_in[1];
    const float* emb = (const float*)d_in[2];
    const float* Wf  = (const float*)d_in[3];
    const float* Uf  = (const float*)d_in[4];
    const float* bfv = (const float*)d_in[5];
    const float* Wb  = (const float*)d_in[6];
    const float* Ub  = (const float*)d_in[7];
    const float* bbv = (const float*)d_in[8];
    const float* W1  = (const float*)d_in[9];
    const float* b1  = (const float*)d_in[10];
    const float* W2  = (const float*)d_in[11];
    const float* b2  = (const float*)d_in[12];
    const float* Wo  = (const float*)d_in[13];
    const float* bo  = (const float*)d_in[14];
    float* out = (float*)d_out;

    char* p = (char*)d_ws;
    size_t off = 0;
    float* embW = (float*)(p + off); off += (size_t)2 * V_ * G4_ * sizeof(float);
    bf16*  UT   = (bf16*) (p + off); off += (size_t)2 * G4_ * U_ * sizeof(bf16);
    bf16*  W1T  = (bf16*) (p + off); off += (size_t)H_ * ZW_ * sizeof(bf16);
    bf16*  W2T  = (bf16*) (p + off); off += (size_t)H_ * H_ * sizeof(bf16);
    bf16*  C1   = (bf16*) (p + off); off += (size_t)B_ * H_ * sizeof(bf16);
    bf16*  C2   = (bf16*) (p + off); off += (size_t)B_ * H_ * sizeof(bf16);
    bf16*  zbuf = (bf16*) (p + off); off += (size_t)B_ * ZW_ * sizeof(bf16);
    (void)ws_size; (void)in_sizes; (void)n_in; (void)out_size;

    // --- prep: LUTs + bf16 transposed weights + bio features ---
    embw_kernel<<<(2 * V_ * G4_ + 255) / 256, 256, 0, stream>>>(emb, Wf, bfv, Wb, bbv, embW);
    transcvt_kernel<<<(G4_ * U_ + 255) / 256, 256, 0, stream>>>(Uf, UT, U_, G4_);
    transcvt_kernel<<<(G4_ * U_ + 255) / 256, 256, 0, stream>>>(Ub, UT + (size_t)G4_ * U_, U_, G4_);
    transcvt_kernel<<<(H_ * ZW_ + 255) / 256, 256, 0, stream>>>(W1, W1T, ZW_, H_);
    transcvt_kernel<<<(H_ * H_ + 255) / 256, 256, 0, stream>>>(W2, W2T, H_, H_);
    bio_kernel<<<(B_ * F_ + 255) / 256, 256, 0, stream>>>(bio, zbuf);

    // --- bidirectional LSTM (fwd+bwd in parallel over grid.y) ---
    dim3 lg(B_ / LROWS, 2);
    lstm_kernel<<<lg, 1024, 0, stream>>>(seq, embW, UT, zbuf);

    // --- MLP ---
    dim3 g1(B_ / 128, H_ / 64);
    gemm_kernel<<<g1, 256, 0, stream>>>(zbuf, W1T, b1, C1, ZW_, H_, 1);
    gemm_kernel<<<g1, 256, 0, stream>>>(C1,   W2T, b2, C2, H_,  H_, 1);

    // --- head ---
    head_kernel<<<B_ / 8, 256, 0, stream>>>(C2, Wo, bo, out);
}
